// EncoderAttentionGqa_33157147525728
// MI455X (gfx1250) — matrix operand order repaired and checked by reading, not compiled, not run on hardware
//
#include <hip/hip_runtime.h>

#define BB   2
#define LL   2048
#define HH   2048
#define NHH  16
#define KVHH 4
#define DD   128
#define KVD  512   // KVH * D

typedef _Float16 h16;
typedef __attribute__((ext_vector_type(8)))  _Float16 v8h;
typedef __attribute__((ext_vector_type(16))) _Float16 v16h;
typedef __attribute__((ext_vector_type(8)))  float    v8f;

union F16x16 { v16h v; v8h h[2]; };

static __device__ __forceinline__ v8f wmma16(v16h a, v16h b, v8f c) {
  // D = A(16x32 f16) * B(32x16 f16) + C(16x16 f32)
  return __builtin_amdgcn_wmma_f32_16x16x32_f16(false, a, false, b, (short)0, c,
                                                false, false);
}

// async copy: 16 bytes per lane, global -> LDS, tracked by ASYNCcnt
static __device__ __forceinline__ void async_g2l_b128(unsigned lds_addr,
                                                      unsigned long long gaddr) {
  asm volatile("global_load_async_to_lds_b128 %0, %1, off"
               :: "v"(lds_addr), "v"(gaddr) : "memory");
}
static __device__ __forceinline__ void wait_asynccnt0() {
  asm volatile("s_wait_asynccnt 0x0" ::: "memory");
}

// ---------------------------------------------------------------- f32 -> f16
__global__ void cvt_kernel(const float* __restrict__ s, h16* __restrict__ d, int n) {
  int i = blockIdx.x * 256 + threadIdx.x;
  if (i < n) d[i] = (h16)s[i];
}

// ------------------------------------------------- GEMM: C[n,o] = A[n,:]·W[o,:] + bias[o]
// A: [N,K] f16 row-major; W: [O,K] f16 row-major (i.e. computes A @ W^T).
// Each wave computes a 16(M) x 64(N) tile; K-chunks of 32 per WMMA.
template<int OUT_F16>
__global__ __launch_bounds__(128)
void gemm_bias_kernel(const h16* __restrict__ A, const h16* __restrict__ W,
                      const float* __restrict__ bias, void* __restrict__ outp,
                      int N, int K, int O)
{
  const int gw   = blockIdx.x * 4 + threadIdx.y;
  const int nt64 = O >> 6;
  const int mt   = gw / nt64;
  const int nt   = gw % nt64;
  if (mt * 16 >= N) return;

  const int lane = threadIdx.x;
  const int rA   = lane & 15;        // A-frag row
  const int hi   = lane >> 4;
  const int kb   = hi * 16;          // A-frag K base (and +8)
  const int klo  = hi * 16;          // B-frag K base
  const int colB = lane & 15;        // B-frag / C-frag column

  const h16* aRow = A + (size_t)(mt * 16 + rA) * K;

  const v8f vzero = {0,0,0,0,0,0,0,0};
  v8f acc[4];
  #pragma unroll
  for (int s = 0; s < 4; s++) acc[s] = vzero;

  for (int kc = 0; kc < K; kc += 32) {
    F16x16 a;
    a.h[0] = *(const v8h*)(aRow + kc + kb);
    a.h[1] = *(const v8h*)(aRow + kc + kb + 8);
    #pragma unroll
    for (int s = 0; s < 4; s++) {
      const h16* wRow = W + (size_t)(nt * 64 + s * 16 + colB) * K + kc + klo;
      F16x16 bm;
      bm.h[0] = *(const v8h*)(wRow);
      bm.h[1] = *(const v8h*)(wRow + 8);
      acc[s] = wmma16(a.v, bm.v, acc[s]);
    }
  }

  const int rowBase = mt * 16 + hi * 8;
  #pragma unroll
  for (int s = 0; s < 4; s++) {
    const int oc = nt * 64 + s * 16 + colB;
    const float bv = bias[oc];
    #pragma unroll
    for (int j = 0; j < 8; j++) {
      float r = acc[s][j] + bv;
      size_t idx = (size_t)(rowBase + j) * O + oc;
      if (OUT_F16) ((h16*)outp)[idx] = (h16)r;
      else         ((float*)outp)[idx] = r;
    }
  }
}

// ------------------------------------------------- flash attention (GQA)
// grid: (L/64, NH, B), block: (32,4). Wave w owns 16 query rows.
__global__ __launch_bounds__(128)
void attn_kernel(const h16* __restrict__ Q, const h16* __restrict__ Km,
                 const h16* __restrict__ Vm, const float* __restrict__ mask,
                 h16* __restrict__ ctx)
{
  __shared__ h16 Ks[32 * 128];      // K block, [key][d]   (filled by async DMA)
  __shared__ h16 Vst[128 * 32];     // V block transposed, [d][key]
  __shared__ h16 Ps[4][16 * 32];    // per-wave P tile, [row][key]

  const int b    = blockIdx.z;
  const int h    = blockIdx.y;
  const int kvh  = h >> 2;          // G = 4
  const int w    = threadIdx.y;
  const int lane = threadIdx.x;
  const int tid  = w * 32 + lane;
  const int qrow0 = blockIdx.x * 64 + w * 16;

  const int rA   = lane & 15;
  const int hi   = lane >> 4;
  const int kb   = hi * 16;
  const int klo  = hi * 16;
  const int colB = lane & 15;

  // load Q fragments once (16 rows x 128 d -> 4 K-chunks of 32)
  const h16* qptr = Q + (size_t)(b * LL + qrow0 + rA) * HH + h * DD;
  F16x16 aq[4];
  #pragma unroll
  for (int kc = 0; kc < 4; kc++) {
    aq[kc].h[0] = *(const v8h*)(qptr + kc * 32 + kb);
    aq[kc].h[1] = *(const v8h*)(qptr + kc * 32 + kb + 8);
  }

  // all-ones B fragment: P @ ones gives the softmax row-sum in every column
  F16x16 ones;
  #pragma unroll
  for (int i = 0; i < 16; i++) ones.v[i] = (h16)1.0f;

  const v8f vzero = {0,0,0,0,0,0,0,0};
  v8f oacc[8];
  #pragma unroll
  for (int dt = 0; dt < 8; dt++) oacc[dt] = vzero;
  v8f lacc = vzero;                 // online softmax denominators (per row)
  float mrow[8];
  #pragma unroll
  for (int j = 0; j < 8; j++) mrow[j] = -1e30f;

  const float scale = 0.08838834764831845f;   // 1/sqrt(128)

  const int keyTid = tid >> 2;      // 0..31
  const int part   = tid & 3;
  const int d0     = part * 32;

  for (int jb = 0; jb < LL / 32; jb++) {
    // ---- stage K via async global->LDS DMA; V loaded + transposed via VGPRs
    {
      const size_t rowoff = (size_t)(b * LL + jb * 32 + keyTid) * KVD + kvh * DD + d0;
      const h16* ksrc = Km + rowoff;
      const h16* vsrc = Vm + rowoff;
      #pragma unroll
      for (int t = 0; t < 4; t++) {
        async_g2l_b128((unsigned)(size_t)&Ks[keyTid * 128 + d0 + t * 8],
                       (unsigned long long)(size_t)(ksrc + t * 8));
        v8h vv = *(const v8h*)(vsrc + t * 8);
        #pragma unroll
        for (int i = 0; i < 8; i++) Vst[(d0 + t * 8 + i) * 32 + keyTid] = vv[i];
      }
    }
    wait_asynccnt0();
    __syncthreads();

    // ---- scores S = Q K^T for two 16-key column blocks
    v8f s0 = vzero, s1 = vzero;
    #pragma unroll
    for (int kc = 0; kc < 4; kc++) {
      F16x16 bk0, bk1;
      bk0.h[0] = *(const v8h*)&Ks[colB * 128 + kc * 32 + klo];
      bk0.h[1] = *(const v8h*)&Ks[colB * 128 + kc * 32 + klo + 8];
      s0 = wmma16(aq[kc].v, bk0.v, s0);
      bk1.h[0] = *(const v8h*)&Ks[(16 + colB) * 128 + kc * 32 + klo];
      bk1.h[1] = *(const v8h*)&Ks[(16 + colB) * 128 + kc * 32 + klo + 8];
      s1 = wmma16(aq[kc].v, bk1.v, s1);
    }

    const float mk0 = mask[b * LL + jb * 32 + colB];
    const float mk1 = mask[b * LL + jb * 32 + 16 + colB];

    // ---- online softmax: row-max via 16-lane shuffles; row-sum via WMMA below
    #pragma unroll
    for (int j = 0; j < 8; j++) {
      float x0 = s0[j] * scale + mk0;
      float x1 = s1[j] * scale + mk1;
      float lm = fmaxf(x0, x1);
      #pragma unroll
      for (int off = 1; off < 16; off <<= 1)
        lm = fmaxf(lm, __shfl_xor(lm, off, 16));
      float nm    = fmaxf(mrow[j], lm);
      float alpha = __expf(mrow[j] - nm);
      float p0    = __expf(x0 - nm);
      float p1    = __expf(x1 - nm);
      mrow[j] = nm;
      lacc[j] *= alpha;
      #pragma unroll
      for (int dt = 0; dt < 8; dt++) oacc[dt][j] *= alpha;
      const int prow = j + hi * 8;
      Ps[w][prow * 32 + colB]      = (h16)p0;
      Ps[w][prow * 32 + 16 + colB] = (h16)p1;
    }

    // ---- O += P @ V ; l += P @ 1   (P reshaped C->A layout via per-wave LDS tile)
    F16x16 ap;
    ap.h[0] = *(const v8h*)&Ps[w][rA * 32 + kb];
    ap.h[1] = *(const v8h*)&Ps[w][rA * 32 + kb + 8];
    lacc = wmma16(ap.v, ones.v, lacc);
    #pragma unroll
    for (int dt = 0; dt < 8; dt++) {
      F16x16 bv;
      bv.h[0] = *(const v8h*)&Vst[(dt * 16 + colB) * 32 + klo];
      bv.h[1] = *(const v8h*)&Vst[(dt * 16 + colB) * 32 + klo + 8];
      oacc[dt] = wmma16(ap.v, bv.v, oacc[dt]);
    }
    __syncthreads();
  }

  // ---- normalize and store context (f16) in b l (h d) layout
  #pragma unroll
  for (int j = 0; j < 8; j++) {
    const float inv = 1.0f / lacc[j];
    const int row = qrow0 + hi * 8 + j;
    h16* optr = ctx + (size_t)(b * LL + row) * HH + h * DD;
    #pragma unroll
    for (int dt = 0; dt < 8; dt++)
      optr[dt * 16 + colB] = (h16)(oacc[dt][j] * inv);
  }
}

// ------------------------------------------------- residual + LayerNorm
__global__ __launch_bounds__(256)
void add_ln_kernel(const float* __restrict__ proj, const float* __restrict__ resid,
                   const float* __restrict__ g, const float* __restrict__ beta,
                   float* __restrict__ out)
{
  __shared__ float rs[256], rq[256];
  const int row = blockIdx.x;
  const int tid = threadIdx.x;
  float y[8];
  float s = 0.f, q = 0.f;
  #pragma unroll
  for (int t = 0; t < 8; t++) {
    int c = tid + t * 256;
    float v = proj[(size_t)row * HH + c] + resid[(size_t)row * HH + c];
    y[t] = v; s += v; q += v * v;
  }
  rs[tid] = s; rq[tid] = q;
  __syncthreads();
  for (int off = 128; off > 0; off >>= 1) {
    if (tid < off) { rs[tid] += rs[tid + off]; rq[tid] += rq[tid + off]; }
    __syncthreads();
  }
  const float mu  = rs[0] / (float)HH;
  const float var = rq[0] / (float)HH - mu * mu;
  const float inv = rsqrtf(var + 1e-12f);
  #pragma unroll
  for (int t = 0; t < 8; t++) {
    int c = tid + t * 256;
    out[(size_t)row * HH + c] = (y[t] - mu) * inv * g[c] + beta[c];
  }
}

// ----------------------------------------------------------------- launch
extern "C" void kernel_launch(void* const* d_in, const int* in_sizes, int n_in,
                              void* d_out, int out_size, void* d_ws, size_t ws_size,
                              hipStream_t stream)
{
  const float* x    = (const float*)d_in[0];
  const float* mask = (const float*)d_in[1];
  const float* Wq   = (const float*)d_in[2];
  const float* bq   = (const float*)d_in[3];
  const float* Wk   = (const float*)d_in[4];
  const float* bk   = (const float*)d_in[5];
  const float* Wv   = (const float*)d_in[6];
  const float* bv   = (const float*)d_in[7];
  const float* Wo   = (const float*)d_in[8];
  const float* bo   = (const float*)d_in[9];
  const float* ln_g = (const float*)d_in[10];
  const float* ln_b = (const float*)d_in[11];
  float* out = (float*)d_out;

  const int NR = BB * LL;   // 4096 token rows

  char* ws = (char*)d_ws;
  size_t off = 0;
  auto alloc = [&](size_t bytes) -> char* {
    char* p = ws + off;
    off += (bytes + 255) & ~(size_t)255;
    return p;
  };
  h16*   xh   = (h16*)  alloc((size_t)NR * HH * 2);
  h16*   wqh  = (h16*)  alloc((size_t)HH * HH * 2);
  h16*   wkh  = (h16*)  alloc((size_t)KVD * HH * 2);
  h16*   wvh  = (h16*)  alloc((size_t)KVD * HH * 2);
  h16*   woh  = (h16*)  alloc((size_t)HH * HH * 2);
  h16*   qh   = (h16*)  alloc((size_t)NR * HH * 2);
  h16*   kh   = (h16*)  alloc((size_t)NR * KVD * 2);
  h16*   vh   = (h16*)  alloc((size_t)NR * KVD * 2);
  h16*   ctx  = (h16*)  alloc((size_t)NR * HH * 2);
  float* proj = (float*)alloc((size_t)NR * HH * 4);

  auto cvt = [&](const float* s, h16* d, int n) {
    cvt_kernel<<<(n + 255) / 256, 256, 0, stream>>>(s, d, n);
  };
  cvt(x,  xh,  NR * HH);
  cvt(Wq, wqh, HH * HH);
  cvt(Wk, wkh, KVD * HH);
  cvt(Wv, wvh, KVD * HH);
  cvt(Wo, woh, HH * HH);

  dim3 blk(32, 4);
  {
    int waves = (NR / 16) * (HH / 64);
    gemm_bias_kernel<1><<<waves / 4, blk, 0, stream>>>(xh, wqh, bq, qh, NR, HH, HH);
  }
  {
    int waves = (NR / 16) * (KVD / 64);
    gemm_bias_kernel<1><<<waves / 4, blk, 0, stream>>>(xh, wkh, bk, kh, NR, HH, KVD);
    gemm_bias_kernel<1><<<waves / 4, blk, 0, stream>>>(xh, wvh, bv, vh, NR, HH, KVD);
  }
  attn_kernel<<<dim3(LL / 64, NHH, BB), blk, 0, stream>>>(qh, kh, vh, mask, ctx);
  {
    int waves = (NR / 16) * (HH / 64);
    gemm_bias_kernel<0><<<waves / 4, blk, 0, stream>>>(ctx, woh, bo, proj, NR, HH, HH);
  }
  add_ln_kernel<<<NR, 256, 0, stream>>>(proj, x, ln_g, ln_b, out);
}
